// MixBlock_68659347194418
// MI455X (gfx1250) — matrix operand order repaired and checked by reading, not compiled, not run on hardware
//
#include <hip/hip_runtime.h>
#include <math.h>

// ---------------------------------------------------------------------------
// MI455X (gfx1250) implementation.
//  - All 7 dense GEMMs (M = 32768) run through v_wmma_f32_16x16x32_bf16,
//    weights pre-transposed to [N,K] bf16 so A and B fragments are contiguous
//    16B loads using k = 16*half + s in both A and B.
//  - Everything with tiny inner dims (RoPE, kv 16x16, selective scan, x_proj)
//    stays VALU; the scan uses 16 lanes per (b,channel) with shfl_xor reduce.
// ---------------------------------------------------------------------------

#define BB 2
#define HH 128
#define WW 128
#define CC 96
#define LL (HH * WW)          // 16384
#define BL (BB * LL)          // 32768
#define NHEADS 6
#define HD 16
#define DSTATE 16
#define DTRANK 6
#define HIDDEN (4 * CC)       // 384

typedef __attribute__((ext_vector_type(16))) __bf16 bf16x16;
typedef __attribute__((ext_vector_type(8)))  float  f32x8;

union ABFrag { bf16x16 v; uint4 q[2]; };

__device__ __forceinline__ unsigned short f2bf(float f) {
    unsigned int u = __float_as_uint(f);
    u += 0x7fffu + ((u >> 16) & 1u);   // round-to-nearest-even
    return (unsigned short)(u >> 16);
}
__device__ __forceinline__ float silu(float x) { return x / (1.0f + expf(-x)); }
__device__ __forceinline__ float elu1(float x) { return x > 0.0f ? x + 1.0f : expf(x); }

// ---------------------------------------------------------------------------
// Weight prep: fp32 [K,N] row-major -> bf16 [N,K] (transposed)
// ---------------------------------------------------------------------------
__global__ void k_transpose_w(const float* __restrict__ in, unsigned short* __restrict__ out,
                              int K, int N) {
    int idx = blockIdx.x * blockDim.x + threadIdx.x;
    if (idx >= K * N) return;
    int n = idx % N, k = idx / N;
    out[(size_t)n * K + k] = f2bf(in[(size_t)k * N + n]);
}

// ---------------------------------------------------------------------------
// LayerNorm over C=96; one wave per row (3 elements / lane), writes bf16
// ---------------------------------------------------------------------------
__global__ void k_layernorm(const float* __restrict__ x, const float* __restrict__ g,
                            const float* __restrict__ b, unsigned short* __restrict__ outb,
                            int rows) {
    int wave = threadIdx.x >> 5, lane = threadIdx.x & 31;
    int row = blockIdx.x * (blockDim.x >> 5) + wave;
    if (row >= rows) return;
    const float* xr = x + (size_t)row * CC;
    float v0 = xr[lane], v1 = xr[lane + 32], v2 = xr[lane + 64];
    float s = v0 + v1 + v2;
    float sq = v0 * v0 + v1 * v1 + v2 * v2;
    #pragma unroll
    for (int m = 16; m >= 1; m >>= 1) { s += __shfl_xor(s, m, 32); sq += __shfl_xor(sq, m, 32); }
    float mean = s * (1.0f / CC);
    float var = sq * (1.0f / CC) - mean * mean;
    float inv = rsqrtf(var + 1e-5f);
    unsigned short* o = outb + (size_t)row * CC;
    o[lane]      = f2bf((v0 - mean) * inv * g[lane]      + b[lane]);
    o[lane + 32] = f2bf((v1 - mean) * inv * g[lane + 32] + b[lane + 32]);
    o[lane + 64] = f2bf((v2 - mean) * inv * g[lane + 64] + b[lane + 64]);
}

// ---------------------------------------------------------------------------
// WMMA GEMM: out[M,N] = A[M,K](bf16) @ Wt[N,K](bf16)^T  (+bias)(+act)(+resid)
// one 16x16 tile per wave, K-step 32, block = 4 waves along M
// ---------------------------------------------------------------------------
template <int ACT>  // 0 = none, 1 = exact GELU
__global__ void k_gemm_wmma(const unsigned short* __restrict__ A,
                            const unsigned short* __restrict__ Wt,
                            int M, int N, int K,
                            const float* __restrict__ bias,
                            const float* __restrict__ resid,
                            float* __restrict__ outF,
                            unsigned short* __restrict__ outB, int ldb, int coloff) {
    int wave = threadIdx.x >> 5, lane = threadIdx.x & 31;
    int tm = blockIdx.y * 4 + wave;
    int tn = blockIdx.x;
    int m0 = tm * 16;
    if (m0 >= M) return;                       // wave-uniform: EXEC stays full
    int half = lane >> 4, r = lane & 15;
    const unsigned short* Arow = A  + (size_t)(m0 + r) * K;
    const unsigned short* Brow = Wt + (size_t)(tn * 16 + r) * K + half * 16;
    f32x8 acc = {};
    for (int k0 = 0; k0 < K; k0 += 32) {
        ABFrag a, bfr;
        a.q[0]   = *(const uint4*)(Arow + k0 + half * 16);
        a.q[1]   = *(const uint4*)(Arow + k0 + half * 16 + 8);
        bfr.q[0] = *(const uint4*)(Brow + k0);
        bfr.q[1] = *(const uint4*)(Brow + k0 + 8);
        if (k0 + 32 < K) {
            __builtin_prefetch((const void*)(Arow + k0 + 32), 0, 1);
            __builtin_prefetch((const void*)(Brow + k0 + 32), 0, 1);
        }
        acc = __builtin_amdgcn_wmma_f32_16x16x32_bf16(false, a.v, false, bfr.v,
                                                      (short)0, acc, false, false);
    }
    int col = tn * 16 + r;
    float bv = bias ? bias[col] : 0.0f;
    #pragma unroll
    for (int i = 0; i < 8; i++) {
        int row = m0 + half * 8 + i;
        float v = acc[i] + bv;
        if (ACT == 1) v = 0.5f * v * (1.0f + erff(v * 0.70710678118654752f));
        if (resid) v += resid[(size_t)row * N + col];
        if (outF) outF[(size_t)row * N + col] = v;
        if (outB) outB[(size_t)row * ldb + coloff + col] = f2bf(v);
    }
}

// ---------------------------------------------------------------------------
// Depthwise 3x3 conv (on w_ slice of xzw, ch 192..287) + SiLU -> lin_x
// ---------------------------------------------------------------------------
__global__ void k_dwconv2d_silu(const float* __restrict__ xzw, const float* __restrict__ wgt,
                                const float* __restrict__ bias,
                                float* __restrict__ outF, unsigned short* __restrict__ outB) {
    int idx = blockIdx.x * blockDim.x + threadIdx.x;
    if (idx >= BL * CC) return;
    int c = idx % CC;
    int l = (idx / CC) % LL;
    int b = idx / (CC * LL);
    int h = l >> 7, w = l & 127;
    float s = bias[c];
    #pragma unroll
    for (int dh = -1; dh <= 1; dh++) {
        int h2 = h + dh;
        if (h2 < 0 || h2 >= HH) continue;
        #pragma unroll
        for (int dw = -1; dw <= 1; dw++) {
            int w2 = w + dw;
            if (w2 < 0 || w2 >= WW) continue;
            s += xzw[((size_t)b * LL + h2 * WW + w2) * 288 + 192 + c] *
                 wgt[c * 9 + (dh + 1) * 3 + (dw + 1)];
        }
    }
    float y = silu(s);
    outF[idx] = y;
    outB[idx] = f2bf(y);
}

// ---------------------------------------------------------------------------
// ELU+1 and RoPE on qk -> q, k, q_r, k_r (one thread per rotation pair)
// ---------------------------------------------------------------------------
__global__ void k_elu_rope(const float* __restrict__ qk,
                           float* __restrict__ q, float* __restrict__ k,
                           float* __restrict__ qr, float* __restrict__ kr) {
    int idx = blockIdx.x * blockDim.x + threadIdx.x;
    if (idx >= BL * 48) return;
    int i = idx % 48;
    int rl = idx / 48;
    int l = rl % LL;
    const float* row = qk + (size_t)rl * 192;
    float q0 = elu1(row[2 * i]),      q1 = elu1(row[2 * i + 1]);
    float k0 = elu1(row[96 + 2 * i]), k1 = elu1(row[96 + 2 * i + 1]);
    int j = (i < 24) ? i : i - 24;
    float pos = (i < 24) ? (float)(l >> 7) : (float)(l & 127);
    float theta = expf(-(float)j * (9.210340371976184f / 24.0f)); // 10000^(-j/24)
    float ang = pos * theta;
    float cs = cosf(ang), sn = sinf(ang);
    size_t o = (size_t)rl * CC + 2 * i;
    q[o] = q0;  q[o + 1] = q1;
    k[o] = k0;  k[o + 1] = k1;
    qr[o] = q0 * cs - q1 * sn;  qr[o + 1] = q0 * sn + q1 * cs;
    kr[o] = k0 * cs - k1 * sn;  kr[o + 1] = k0 * sn + k1 * cs;
}

// kmean[b,c] = mean over L of k[b,l,c]
__global__ void k_kmean(const float* __restrict__ k, float* __restrict__ kmean) {
    int c = blockIdx.x, b = blockIdx.y, t = threadIdx.x;
    __shared__ float red[256];
    float s = 0.0f;
    for (int l = t; l < LL; l += 256) s += k[((size_t)b * LL + l) * CC + c];
    red[t] = s;
    __syncthreads();
    for (int st = 128; st >= 1; st >>= 1) {
        if (t < st) red[t] += red[t + st];
        __syncthreads();
    }
    if (t == 0) kmean[b * CC + c] = red[0] * (1.0f / LL);
}

// kv[b,h,d,e] = (1/L) * sum_l k_r[b,l,h*16+d] * v[b,l,h*16+e]
__global__ void k_kv(const float* __restrict__ kr, const float* __restrict__ v,
                     float* __restrict__ kv) {
    int d = blockIdx.x, h = blockIdx.y, b = blockIdx.z;
    int t = threadIdx.x;
    int e = t & 15, chunk = t >> 4;
    __shared__ float red[256];
    float s = 0.0f;
    for (int l = chunk; l < LL; l += 16) {
        size_t base = ((size_t)b * LL + l) * CC + h * HD;
        s += kr[base + d] * v[base + e];
    }
    red[t] = s;
    __syncthreads();
    for (int st = 128; st >= 16; st >>= 1) {
        if (t < st) red[t] += red[t + st];
        __syncthreads();
    }
    if (t < 16) kv[(((size_t)(b * NHEADS + h) * HD) + d) * HD + t] = red[t] * (1.0f / LL);
}

// attn = (q_r @ kv) * zden + lepe(dwconv2d(v))
__global__ void k_attn_out(const float* __restrict__ q, const float* __restrict__ qr,
                           const float* __restrict__ kmean, const float* __restrict__ kv,
                           const float* __restrict__ linx,
                           const float* __restrict__ lepe_w, const float* __restrict__ lepe_b,
                           float* __restrict__ attn) {
    int idx = blockIdx.x * blockDim.x + threadIdx.x;
    if (idx >= BL * CC) return;
    int c = idx % CC;
    int l = (idx / CC) % LL;
    int b = idx / (CC * LL);
    size_t rl = (size_t)b * LL + l;
    int hid = c >> 4, e = c & 15;
    float zs = 0.0f, o = 0.0f;
    const float* kvp = kv + ((size_t)(b * NHEADS + hid) * HD) * HD;
    #pragma unroll
    for (int d = 0; d < HD; d++) {
        zs += q[rl * CC + hid * HD + d] * kmean[b * CC + hid * HD + d];
        o  += qr[rl * CC + hid * HD + d] * kvp[d * HD + e];
    }
    float zden = 1.0f / (zs + 1e-6f);
    int h = l >> 7, w = l & 127;
    float s = lepe_b[c];
    #pragma unroll
    for (int dh = -1; dh <= 1; dh++) {
        int h2 = h + dh;
        if (h2 < 0 || h2 >= HH) continue;
        #pragma unroll
        for (int dw = -1; dw <= 1; dw++) {
            int w2 = w + dw;
            if (w2 < 0 || w2 >= WW) continue;
            s += linx[((size_t)b * LL + h2 * WW + w2) * CC + c] *
                 lepe_w[c * 9 + (dh + 1) * 3 + (dw + 1)];
        }
    }
    attn[idx] = o * zden + s;
}

// ---------------------------------------------------------------------------
// Depthwise 1-D conv (k=4, pad left 1 right 2) + SiLU on a channel slice of xzw
// ---------------------------------------------------------------------------
__global__ void k_conv1d_silu(const float* __restrict__ xzw, int choff,
                              const float* __restrict__ w4,
                              float* __restrict__ outF,
                              unsigned short* __restrict__ outB, int ldb, int coloff) {
    int idx = blockIdx.x * blockDim.x + threadIdx.x;
    if (idx >= BL * CC) return;
    int c = idx % CC;
    int l = (idx / CC) % LL;
    int b = idx / (CC * LL);
    float s = 0.0f;
    #pragma unroll
    for (int j = 0; j < 4; j++) {
        int li = l + j - 1;
        if (li >= 0 && li < LL)
            s += xzw[((size_t)b * LL + li) * 288 + choff + c] * w4[c * 4 + j];
    }
    float y = silu(s);
    size_t rl = (size_t)b * LL + l;
    outF[rl * CC + c] = y;
    if (outB) outB[rl * ldb + coloff + c] = f2bf(y);
}

// x_dbl[b,l,e] = sum_d x_c[b,l,d] * x_proj_w[d,e], e < 38
__global__ void k_xdbl(const float* __restrict__ xc, const float* __restrict__ xpw,
                       float* __restrict__ xdbl) {
    int idx = blockIdx.x * blockDim.x + threadIdx.x;
    if (idx >= BL * 38) return;
    int e = idx % 38;
    size_t rl = idx / 38;
    float s = 0.0f;
    for (int d = 0; d < CC; d++) s += xc[rl * CC + d] * xpw[d * 38 + e];
    xdbl[idx] = s;
}

// delta[b,l,c] = softplus( x_dbl[b,l,:6] @ dt_proj_w[:,c] + dt_proj_b[c] )
__global__ void k_delta(const float* __restrict__ xdbl, const float* __restrict__ dtw,
                        const float* __restrict__ dtb, float* __restrict__ delta) {
    int idx = blockIdx.x * blockDim.x + threadIdx.x;
    if (idx >= BL * CC) return;
    int c = idx % CC;
    size_t rl = idx / CC;
    float s = dtb[c];
    #pragma unroll
    for (int r = 0; r < DTRANK; r++) s += xdbl[rl * 38 + r] * dtw[r * CC + c];
    delta[idx] = (s > 20.0f) ? s : log1pf(expf(s));
}

// selective scan: 16 lanes (states) per (b, channel); sequential over L
__global__ void k_scan(const float* __restrict__ delta, const float* __restrict__ xc,
                       const float* __restrict__ xdbl, const float* __restrict__ Alog,
                       const float* __restrict__ Dv, unsigned short* __restrict__ yz) {
    int lane = threadIdx.x;
    int pair = blockIdx.x * 2 + (lane >> 4);
    int b = pair / CC, d = pair % CC;
    int n = lane & 15;
    float A_dn = -expf(Alog[d * DSTATE + n]);
    float Dd = Dv[d];
    float h = 0.0f;
    for (int l = 0; l < LL; l++) {
        size_t rl = (size_t)b * LL + l;
        float dl = delta[rl * CC + d];
        float u  = xc[rl * CC + d];
        float Bv = xdbl[rl * 38 + DTRANK + n];
        float Cv = xdbl[rl * 38 + DTRANK + DSTATE + n];
        h = expf(dl * A_dn) * h + dl * Bv * u;
        float p = h * Cv;
        p += __shfl_xor(p, 1, 32);
        p += __shfl_xor(p, 2, 32);
        p += __shfl_xor(p, 4, 32);
        p += __shfl_xor(p, 8, 32);
        if (n == 0) yz[rl * 192 + d] = f2bf(p + Dd * u);
    }
}

// out2 GEMM A input: bf16(attn * z)
__global__ void k_mul_bf(const float* __restrict__ a, const float* __restrict__ b,
                         unsigned short* __restrict__ o) {
    int idx = blockIdx.x * blockDim.x + threadIdx.x;
    if (idx >= BL * CC) return;
    o[idx] = f2bf(a[idx] * b[idx]);
}

// ---------------------------------------------------------------------------
static inline int cdiv(int a, int b) { return (a + b - 1) / b; }

extern "C" void kernel_launch(void* const* d_in, const int* in_sizes, int n_in,
                              void* d_out, int out_size, void* d_ws, size_t ws_size,
                              hipStream_t stream) {
    const float* hidden    = (const float*)d_in[0];
    const float* norm_in_g = (const float*)d_in[1];
    const float* norm_in_b = (const float*)d_in[2];
    const float* in_proj_w = (const float*)d_in[3];
    const float* dw_w      = (const float*)d_in[4];
    const float* dw_b      = (const float*)d_in[5];
    const float* qk_w      = (const float*)d_in[6];
    const float* qk_b      = (const float*)d_in[7];
    const float* lepe_w    = (const float*)d_in[8];
    const float* lepe_b    = (const float*)d_in[9];
    const float* conv_x_w  = (const float*)d_in[10];
    const float* conv_z_w  = (const float*)d_in[11];
    const float* x_proj_w  = (const float*)d_in[12];
    const float* dt_proj_w = (const float*)d_in[13];
    const float* dt_proj_b = (const float*)d_in[14];
    const float* A_log     = (const float*)d_in[15];
    const float* Dv        = (const float*)d_in[16];
    const float* out_proj_w= (const float*)d_in[17];
    const float* proj_out_w= (const float*)d_in[18];
    const float* proj_out_b= (const float*)d_in[19];
    const float* out_w     = (const float*)d_in[20];
    const float* out_b     = (const float*)d_in[21];
    const float* norm_mlp_g= (const float*)d_in[22];
    const float* norm_mlp_b= (const float*)d_in[23];
    const float* fc1_w     = (const float*)d_in[24];
    const float* fc1_b     = (const float*)d_in[25];
    const float* fc2_w     = (const float*)d_in[26];
    const float* fc2_b     = (const float*)d_in[27];
    float* out_final = (float*)d_out;

    // ---- workspace carving (256B aligned) ----
    char* ws = (char*)d_ws;
    size_t off = 0;
    auto alloc = [&](size_t bytes) -> char* {
        char* p = ws + off;
        off += (bytes + 255) & ~(size_t)255;
        return p;
    };
    unsigned short* wt_inproj  = (unsigned short*)alloc(288 * 96 * 2);
    unsigned short* wt_qk      = (unsigned short*)alloc(192 * 96 * 2);
    unsigned short* wt_outproj = (unsigned short*)alloc(96 * 192 * 2);
    unsigned short* wt_projout = (unsigned short*)alloc(96 * 96 * 2);
    unsigned short* wt_out     = (unsigned short*)alloc(96 * 192 * 2);
    unsigned short* wt_fc1     = (unsigned short*)alloc(384 * 96 * 2);
    unsigned short* wt_fc2     = (unsigned short*)alloc(96 * 384 * 2);

    unsigned short* hs_bf   = (unsigned short*)alloc((size_t)BL * 96 * 2);
    float*          xzw     = (float*)alloc((size_t)BL * 288 * 4);
    float*          lin_x   = (float*)alloc((size_t)BL * 96 * 4);
    unsigned short* lin_xbf = (unsigned short*)alloc((size_t)BL * 96 * 2);
    float*          qkbuf   = (float*)alloc((size_t)BL * 192 * 4);
    float*          qbuf    = (float*)alloc((size_t)BL * 96 * 4);
    float*          kbuf    = (float*)alloc((size_t)BL * 96 * 4);
    float*          qrbuf   = (float*)alloc((size_t)BL * 96 * 4);
    float*          krbuf   = (float*)alloc((size_t)BL * 96 * 4);
    float*          kmean   = (float*)alloc(BB * 96 * 4);
    float*          kv      = (float*)alloc(BB * NHEADS * HD * HD * 4);
    float*          attn    = (float*)alloc((size_t)BL * 96 * 4);
    float*          x_c     = (float*)alloc((size_t)BL * 96 * 4);
    float*          z_c     = (float*)alloc((size_t)BL * 96 * 4);
    float*          xdbl    = (float*)alloc((size_t)BL * 38 * 4);
    float*          delta   = (float*)alloc((size_t)BL * 96 * 4);
    unsigned short* yz_bf   = (unsigned short*)alloc((size_t)BL * 192 * 2);
    unsigned short* o2a_bf  = (unsigned short*)alloc((size_t)BL * 96 * 2);
    unsigned short* cat2_bf = (unsigned short*)alloc((size_t)BL * 192 * 2);
    float*          out_buf = (float*)alloc((size_t)BL * 96 * 4);
    unsigned short* h2_bf   = (unsigned short*)alloc((size_t)BL * 96 * 2);
    unsigned short* h2a_bf  = (unsigned short*)alloc((size_t)BL * 384 * 2);

    // ---- weight prep (fp32 [K,N] -> bf16 [N,K]) ----
    auto tw = [&](const float* in, unsigned short* out, int K, int N) {
        k_transpose_w<<<cdiv(K * N, 256), 256, 0, stream>>>(in, out, K, N);
    };
    tw(in_proj_w,  wt_inproj,  96, 288);
    tw(qk_w,       wt_qk,      96, 192);
    tw(out_proj_w, wt_outproj, 192, 96);
    tw(proj_out_w, wt_projout, 96, 96);
    tw(out_w,      wt_out,     192, 96);
    tw(fc1_w,      wt_fc1,     96, 384);
    tw(fc2_w,      wt_fc2,     384, 96);

    dim3 gblk(128);
    auto gemm = [&](const unsigned short* A, const unsigned short* Wt, int N, int K,
                    const float* bias, const float* resid, float* oF,
                    unsigned short* oB, int ldb, int coloff, int act) {
        dim3 grid(N / 16, BL / 64);
        if (act == 1)
            k_gemm_wmma<1><<<grid, gblk, 0, stream>>>(A, Wt, BL, N, K, bias, resid, oF, oB, ldb, coloff);
        else
            k_gemm_wmma<0><<<grid, gblk, 0, stream>>>(A, Wt, BL, N, K, bias, resid, oF, oB, ldb, coloff);
    };

    // 1) LN -> hs_bf ; xzw = hs @ in_proj_w   [BL,288]
    k_layernorm<<<BL / 8, 256, 0, stream>>>(hidden, norm_in_g, norm_in_b, hs_bf, BL);
    gemm(hs_bf, wt_inproj, 288, 96, nullptr, nullptr, xzw, nullptr, 0, 0, 0);

    // 2) lin_x = silu(dwconv2d(w_))  (channels 192..287 of xzw)
    k_dwconv2d_silu<<<cdiv(BL * 96, 256), 256, 0, stream>>>(xzw, dw_w, dw_b, lin_x, lin_xbf);

    // 3) qk = lin_x @ qk_w + qk_b ; ELU+1 ; RoPE
    gemm(lin_xbf, wt_qk, 192, 96, qk_b, nullptr, qkbuf, nullptr, 0, 0, 0);
    k_elu_rope<<<cdiv(BL * 48, 256), 256, 0, stream>>>(qkbuf, qbuf, kbuf, qrbuf, krbuf);
    k_kmean<<<dim3(96, BB), 256, 0, stream>>>(kbuf, kmean);
    k_kv<<<dim3(HD, NHEADS, BB), 256, 0, stream>>>(krbuf, lin_x, kv);

    // 4) SSM branch: conv1d+silu on x (ch 0..95) and z (ch 96..191)
    k_conv1d_silu<<<cdiv(BL * 96, 256), 256, 0, stream>>>(xzw, 0,  conv_x_w, x_c, nullptr, 0, 0);
    k_conv1d_silu<<<cdiv(BL * 96, 256), 256, 0, stream>>>(xzw, 96, conv_z_w, z_c, yz_bf, 192, 96);
    k_xdbl<<<cdiv(BL * 38, 256), 256, 0, stream>>>(x_c, x_proj_w, xdbl);
    k_delta<<<cdiv(BL * 96, 256), 256, 0, stream>>>(xdbl, dt_proj_w, dt_proj_b, delta);
    k_scan<<<96, 32, 0, stream>>>(delta, x_c, xdbl, A_log, Dv, yz_bf);

    // 5) attention output (+lepe) and out2 GEMM input
    k_attn_out<<<cdiv(BL * 96, 256), 256, 0, stream>>>(qbuf, qrbuf, kmean, kv, lin_x,
                                                       lepe_w, lepe_b, attn);
    k_mul_bf<<<cdiv(BL * 96, 256), 256, 0, stream>>>(attn, z_c, o2a_bf);

    // 6) out1 = [y|z] @ out_proj_w -> cat2[:, :96]
    gemm(yz_bf, wt_outproj, 96, 192, nullptr, nullptr, nullptr, cat2_bf, 192, 0, 0);
    // 7) out2 = (attn*z) @ proj_out_w + b -> cat2[:, 96:]
    gemm(o2a_bf, wt_projout, 96, 96, proj_out_b, nullptr, nullptr, cat2_bf, 192, 96, 0);
    // 8) out = cat2 @ out_w + out_b + shortcut
    gemm(cat2_bf, wt_out, 96, 192, out_b, hidden, out_buf, nullptr, 0, 0, 0);

    // 9) MLP: LN -> fc1(GELU) -> fc2 (+out residual) -> d_out
    k_layernorm<<<BL / 8, 256, 0, stream>>>(out_buf, norm_mlp_g, norm_mlp_b, h2_bf, BL);
    gemm(h2_bf, wt_fc1, HIDDEN, 96, fc1_b, nullptr, nullptr, h2a_bf, HIDDEN, 0, 1);
    gemm(h2a_bf, wt_fc2, 96, HIDDEN, fc2_b, out_buf, out_final, nullptr, 0, 0, 0);
}
